// DeformableConv2d_31851477467328
// MI455X (gfx1250) — matrix operand order repaired and checked by reading, not compiled, not run on hardware
//
#include <hip/hip_runtime.h>

typedef __attribute__((ext_vector_type(16))) _Float16 v16h;
typedef __attribute__((ext_vector_type(8)))  float    v8f;

#define BATCH 8
#define CIN   64
#define HH    128
#define WW    128
#define K2    9
#define KDIM  576            // K2 * CIN, ordered k2*64 + c
#define NSLAB 18             // KDIM / 32
#define HWSZ  (HH*WW)        // 16384

// ---------------------------------------------------------------------------
// Workspace layout (bytes):
//   [0)         offset : 8*18*128*128 f32 = 9437184 B
//   [9437184)   mask   : 8* 9*128*128 f32 = 4718592 B
//   [14155776)  A2 f16 : 64*576           =   73728 B
//   [14229504)  A1 f16 : 32*576           =   36864 B
#define WS_OFF_OFFSET 0
#define WS_OFF_MASK   9437184
#define WS_OFF_A2     14155776
#define WS_OFF_A1     14229504

// Async copy of one 16B chunk from global to LDS (ASYNCcnt-tracked DMA).
__device__ __forceinline__ void async_b128_to_lds(const _Float16* gsrc,
                                                  _Float16* lds_dst) {
  unsigned laddr = (unsigned)(size_t)lds_dst;   // low 32 bits = LDS offset
  asm volatile("global_load_async_to_lds_b128 %0, %1, off"
               :: "v"(laddr), "v"(gsrc) : "memory");
}
__device__ __forceinline__ void wait_async0() {
  asm volatile("s_wait_asynccnt 0x0" ::: "memory");
}

// ---------------------------------------------------------------------------
// Repack weights to f16 GEMM A-matrices with K = k2*64 + c.
__global__ __launch_bounds__(256) void prep_w(const float* __restrict__ wc,
                                              const float* __restrict__ wo,
                                              const float* __restrict__ wm,
                                              _Float16* __restrict__ A2,
                                              _Float16* __restrict__ A1) {
  int t = blockIdx.x * 256 + threadIdx.x;
  if (t < 64 * KDIM) {                      // A2: 64 x 576
    int o = t / KDIM, k = t - o * KDIM;
    int k2 = k >> 6, c = k & 63;
    A2[t] = (_Float16)wc[(o * 64 + c) * 9 + k2];
  }
  int t2 = t - 64 * KDIM;
  if (t2 >= 0 && t2 < 32 * KDIM) {          // A1: 32 x 576 (rows 27..31 zero)
    int r = t2 / KDIM, k = t2 - r * KDIM;
    int k2 = k >> 6, c = k & 63;
    float v = 0.0f;
    if (r < 18)      v = wo[(r * 64 + c) * 9 + k2];
    else if (r < 27) v = wm[((r - 18) * 64 + c) * 9 + k2];
    A1[t2] = (_Float16)v;
  }
}

// ---------------------------------------------------------------------------
// Stage 1: offset (18ch) and sigmoid(mask) (9ch) via implicit-GEMM WMMA conv.
// One block per (b,h): 128 pixels, 8 waves x N=16, 2 M-tiles (32 rows).
__global__ __launch_bounds__(256) void conv_om(const float* __restrict__ x,
                                               const _Float16* __restrict__ A1,
                                               float* __restrict__ off,
                                               float* __restrict__ msk) {
  __shared__ _Float16 Bl[128][32];   // 8 KB B tile
  __shared__ _Float16 Asl[32 * 32];  // 2 KB A slab (32 rows x K=32)

  const int tid  = threadIdx.x;
  const int bid  = blockIdx.x;
  const int b    = bid >> 7;
  const int h    = bid & 127;
  const int wave = tid >> 5;
  const int lane = tid & 31;
  const int nloc = lane & 15;
  const int khalf = lane >> 4;   // 0: lanes 0-15, 1: lanes 16-31

  const v8f zero = {0.f, 0.f, 0.f, 0.f, 0.f, 0.f, 0.f, 0.f};
  v8f acc0 = zero, acc1 = zero;

  const int fillpx = tid >> 5;   // 0..7 (8 px rows per sweep)
  const int fillcc = tid & 31;

  for (int slab = 0; slab < NSLAB; ++slab) {
    const int k2 = slab >> 1;
    const int cbase = (slab & 1) << 5;
    const int ki = k2 / 3, kj = k2 - ki * 3;
    const int yy = h + ki - 1;
    const bool yok = (yy >= 0) & (yy < HH);

    // Kick off async DMA of the A slab (32 rows x 32 K = 2KB, 128 chunks).
    if (tid < 128) {
      int row = tid >> 2, chunk = tid & 3;
      async_b128_to_lds(A1 + row * KDIM + slab * 32 + chunk * 8,
                        &Asl[row * 32 + chunk * 8]);
    }

    // im2col slab -> LDS (f16), zero-padded borders (overlaps the DMA)
    const float* xc = x + (size_t)(b * CIN + cbase + fillcc) * HWSZ + yy * WW;
    for (int i = 0; i < 16; ++i) {
      int px = i * 8 + fillpx;
      int xx = px + kj - 1;
      float v = 0.0f;
      if (yok && xx >= 0 && xx < WW) v = xc[xx];
      Bl[px][fillcc] = (_Float16)v;
    }
    wait_async0();
    __syncthreads();

    // B fragment: lanes 0-15 K=0..15, lanes 16-31 K=16..31 (pairs per VGPR)
    v16h bfrag;
    const _Float16* bp = &Bl[wave * 16 + nloc][khalf * 16];
    *(uint4*)&bfrag       = *(const uint4*)bp;
    *((uint4*)&bfrag + 1) = *(const uint4*)(bp + 8);

    // A fragments from LDS: lanes 0-15 K 0..15; lanes 16-31 K 16..31
    {
      v16h afrag;
      const _Float16* ap = &Asl[(0 * 16 + nloc) * 32 + khalf * 16];
      *(uint4*)&afrag       = *(const uint4*)ap;
      *((uint4*)&afrag + 1) = *(const uint4*)(ap + 8);
      acc0 = __builtin_amdgcn_wmma_f32_16x16x32_f16(false, afrag, false, bfrag,
                                                    (short)0, acc0, false, false);
    }
    {
      v16h afrag;
      const _Float16* ap = &Asl[(1 * 16 + nloc) * 32 + khalf * 16];
      *(uint4*)&afrag       = *(const uint4*)ap;
      *((uint4*)&afrag + 1) = *(const uint4*)(ap + 8);
      acc1 = __builtin_amdgcn_wmma_f32_16x16x32_f16(false, afrag, false, bfrag,
                                                    (short)0, acc1, false, false);
    }
    __syncthreads();
  }

  // Epilogue: D layout — VGPR r: M = r + 8*khalf, N = nloc
  const int px = wave * 16 + nloc;
  for (int mt = 0; mt < 2; ++mt) {
    for (int r = 0; r < 8; ++r) {
      int m = mt * 16 + r + khalf * 8;
      float v = (mt == 0) ? acc0[r] : acc1[r];
      if (m < 18) {
        off[((size_t)(b * 18 + m) * HH + h) * WW + px] = v;
      } else if (m < 27) {
        msk[((size_t)(b * 9 + (m - 18)) * HH + h) * WW + px] =
            1.0f / (1.0f + __expf(-v));
      }
    }
  }
}

// ---------------------------------------------------------------------------
// Stage 2: bilinear gather (f32 geometry, mask folded into corner weights)
// + WMMA contraction over K=576. One block per (b,h) row: 128 px, M=64.
__global__ __launch_bounds__(256) void deform_main(const float* __restrict__ x,
                                                   const _Float16* __restrict__ A2,
                                                   const float* __restrict__ off,
                                                   const float* __restrict__ msk,
                                                   float* __restrict__ out) {
  __shared__ _Float16 Bl[128][32];       //  8 KB
  __shared__ _Float16 Asl[64 * 32];      //  4 KB A slab (64 rows x K=32)
  __shared__ int      sIdx[1152][4];     // 18 KB  (128 px * 9 taps)
  __shared__ float    sWt[1152][4];      // 18 KB

  const int tid  = threadIdx.x;
  const int bid  = blockIdx.x;
  const int b    = bid >> 7;
  const int h    = bid & 127;
  const int wave = tid >> 5;
  const int lane = tid & 31;
  const int nloc = lane & 15;
  const int khalf = lane >> 4;

  // Prologue: corner indices + mask-premultiplied bilinear weights.
  for (int t = tid; t < 1152; t += 256) {
    int px = t / 9, k2 = t - px * 9;
    int ki = k2 / 3, kj = k2 - ki * 3;
    float oy = off[((size_t)(b * 18 + 2 * k2 + 0) * HH + h) * WW + px];
    float ox = off[((size_t)(b * 18 + 2 * k2 + 1) * HH + h) * WW + px];
    float mk = msk[((size_t)(b * 9 + k2) * HH + h) * WW + px];
    float py  = (float)h  + (float)(ki - 1) + oy;
    float pxf = (float)px + (float)(kj - 1) + ox;
    float y0 = floorf(py), x0 = floorf(pxf);
    float wy1 = py - y0, wy0 = 1.0f - wy1;
    float wx1 = pxf - x0, wx0 = 1.0f - wx1;
    int y0i = (int)y0, x0i = (int)x0;
    int y1i = y0i + 1, x1i = x0i + 1;
    float vy0 = (y0i >= 0 && y0i < HH) ? 1.0f : 0.0f;
    float vy1 = (y1i >= 0 && y1i < HH) ? 1.0f : 0.0f;
    float vx0 = (x0i >= 0 && x0i < WW) ? 1.0f : 0.0f;
    float vx1 = (x1i >= 0 && x1i < WW) ? 1.0f : 0.0f;
    int y0c = min(max(y0i, 0), HH - 1), y1c = min(max(y1i, 0), HH - 1);
    int x0c = min(max(x0i, 0), WW - 1), x1c = min(max(x1i, 0), WW - 1);
    sIdx[t][0] = y0c * WW + x0c;
    sIdx[t][1] = y0c * WW + x1c;
    sIdx[t][2] = y1c * WW + x0c;
    sIdx[t][3] = y1c * WW + x1c;
    sWt[t][0] = wy0 * wx0 * vy0 * vx0 * mk;
    sWt[t][1] = wy0 * wx1 * vy0 * vx1 * mk;
    sWt[t][2] = wy1 * wx0 * vy1 * vx0 * mk;
    sWt[t][3] = wy1 * wx1 * vy1 * vx1 * mk;
  }
  __syncthreads();

  const v8f zero = {0.f, 0.f, 0.f, 0.f, 0.f, 0.f, 0.f, 0.f};
  v8f acc[4] = {zero, zero, zero, zero};

  const int fillpx = tid >> 5;
  const int fillcc = tid & 31;

  for (int slab = 0; slab < NSLAB; ++slab) {
    const int k2 = slab >> 1;
    const int cbase = (slab & 1) << 5;

    // Kick off async DMA of A slab (64 rows x 32 K = 4KB, 256 chunks).
    {
      int row = tid >> 2, chunk = tid & 3;
      async_b128_to_lds(A2 + row * KDIM + slab * 32 + chunk * 8,
                        &Asl[row * 32 + chunk * 8]);
    }

    // Gather + mask-weight -> f16 B tile in LDS (overlaps the DMA).
    const float* xb = x + (size_t)(b * CIN + cbase + fillcc) * HWSZ;
    for (int i = 0; i < 16; ++i) {
      int px = i * 8 + fillpx;
      int t = px * 9 + k2;
      float v = sWt[t][0] * xb[sIdx[t][0]] + sWt[t][1] * xb[sIdx[t][1]] +
                sWt[t][2] * xb[sIdx[t][2]] + sWt[t][3] * xb[sIdx[t][3]];
      Bl[px][fillcc] = (_Float16)v;
    }
    wait_async0();
    __syncthreads();

    v16h bfrag;
    const _Float16* bp = &Bl[wave * 16 + nloc][khalf * 16];
    *(uint4*)&bfrag       = *(const uint4*)bp;
    *((uint4*)&bfrag + 1) = *(const uint4*)(bp + 8);

#pragma unroll
    for (int mt = 0; mt < 4; ++mt) {
      v16h afrag;
      const _Float16* ap = &Asl[(mt * 16 + nloc) * 32 + khalf * 16];
      *(uint4*)&afrag       = *(const uint4*)ap;
      *((uint4*)&afrag + 1) = *(const uint4*)(ap + 8);
      acc[mt] = __builtin_amdgcn_wmma_f32_16x16x32_f16(
          false, afrag, false, bfrag, (short)0, acc[mt], false, false);
    }
    __syncthreads();
  }

  // Epilogue: out (8,64,128,128)
  const int px = wave * 16 + nloc;
#pragma unroll
  for (int mt = 0; mt < 4; ++mt) {
#pragma unroll
    for (int r = 0; r < 8; ++r) {
      int m = mt * 16 + r + khalf * 8;
      out[((size_t)(b * CIN + m) * HH + h) * WW + px] = acc[mt][r];
    }
  }
}

// ---------------------------------------------------------------------------
extern "C" void kernel_launch(void* const* d_in, const int* in_sizes, int n_in,
                              void* d_out, int out_size, void* d_ws, size_t ws_size,
                              hipStream_t stream) {
  (void)in_sizes; (void)n_in; (void)out_size; (void)ws_size;
  const float* x  = (const float*)d_in[0];   // (8,64,128,128)
  const float* wc = (const float*)d_in[1];   // (64,64,3,3)
  const float* wo = (const float*)d_in[2];   // (18,64,3,3)
  const float* wm = (const float*)d_in[3];   // (9,64,3,3)
  float* out = (float*)d_out;

  char* ws = (char*)d_ws;
  float*    off = (float*)(ws + WS_OFF_OFFSET);
  float*    msk = (float*)(ws + WS_OFF_MASK);
  _Float16* A2  = (_Float16*)(ws + WS_OFF_A2);
  _Float16* A1  = (_Float16*)(ws + WS_OFF_A1);

  // 1) weight repack to f16 GEMM layout
  prep_w<<<(96 * KDIM + 255) / 256, 256, 0, stream>>>(wc, wo, wm, A2, A1);
  // 2) offset + sigmoid(mask) convs (WMMA implicit GEMM)
  conv_om<<<BATCH * HH, 256, 0, stream>>>(x, A1, off, msk);
  // 3) deformable gather + WMMA contraction
  deform_main<<<BATCH * HH, 256, 0, stream>>>(x, A2, off, msk, out);
}
